// HierarchicalLFQHVQVAE_31052613550674
// MI455X (gfx1250) — compile-verified, matrix operand order checked by reading, not run on hardware
//
#include <hip/hip_runtime.h>
#include <hip/hip_bf16.h>
#include <cstdint>

#define B_    32768
#define FEAT  512
#define ZD    256
#define QD    128
#define NZC   4096
#define NQC   2048
#define HID   512
#define BOT   64

typedef __attribute__((ext_vector_type(16))) _Float16 v16h;
typedef __attribute__((ext_vector_type(8)))  _Float16 v8h;
typedef __attribute__((ext_vector_type(8)))  float    v8f;

// ---------------- device helpers ----------------

__device__ __forceinline__ float gelu_erf(float x) {
    return 0.5f * x * (1.0f + erff(x * 0.70710678118654752440f));
}
__device__ __forceinline__ float sigm(float x) {
    return 1.0f / (1.0f + expf(-x));
}

// Load a 16-bit WMMA A/B fragment for one lane.
// Layout (ISA 7.12.2, 16-bit A 16x32): lane = (half)*16 + row; halves 0..7 hold
// K = kb+0..7, halves 8..15 hold K = kb+16..23, kb = (lane>>4)*8.
// p points at row_base + k (16B aligned).
__device__ __forceinline__ v16h frag_ld(const _Float16* p, int kb) {
    v8h lo = *(const v8h*)(p + kb);
    v8h hi = *(const v8h*)(p + kb + 16);
    v16h f;
#pragma unroll
    for (int i = 0; i < 8; ++i) { f[i] = lo[i]; f[i + 8] = hi[i]; }
    return f;
}

__device__ __forceinline__ v8f wmma_f16(v16h a, v16h b, v8f c) {
    return __builtin_amdgcn_wmma_f32_16x16x32_f16(false, a, false, b, (short)0, c,
                                                  false, false);
}

// CDNA5 async memory->LDS copy (16B per lane), tracked by ASYNCcnt.
// lds_addr is the wave-relative LDS byte offset (shared array is at offset 0).
__device__ __forceinline__ void async_ld_lds_b128(unsigned lds_addr, const void* g) {
    asm volatile("global_load_async_to_lds_b128 %0, %1, off"
                 :
                 : "v"(lds_addr), "v"((unsigned long long)(uintptr_t)g)
                 : "memory");
}
__device__ __forceinline__ void wait_async0() {
    asm volatile("s_wait_asynccnt 0x0" ::: "memory");
}

__device__ __forceinline__ float block_reduce256(float v, float* sm) {
    const int tid = threadIdx.x;
    sm[tid] = v;
    __syncthreads();
#pragma unroll
    for (int s = 128; s > 0; s >>= 1) {
        if (tid < s) sm[tid] += sm[tid + s];
        __syncthreads();
    }
    return sm[0];
}

// ---------------- small prep kernels ----------------

__global__ void zero4_kernel(float* a) {
    if (threadIdx.x < 4) a[threadIdx.x] = 0.0f;
}

// scale[row] = min(1, softplus(ci[row]) / sum(|W[row,:]|))
__global__ __launch_bounds__(256) void lip_scale_kernel(
    const float* __restrict__ W, const float* __restrict__ ci,
    float* __restrict__ scale, int cols) {
    __shared__ float sm[256];
    const int row = blockIdx.x;
    const float* r = W + (size_t)row * cols;
    float s = 0.f;
    for (int c = threadIdx.x; c < cols; c += 256) s += fabsf(r[c]);
    float tot = block_reduce256(s, sm);
    if (threadIdx.x == 0) {
        float sp = log1pf(expf(ci[row]));  // softplus
        scale[row] = fminf(1.0f, sp / tot);
    }
}

__global__ __launch_bounds__(256) void cb_norm_kernel(
    const float* __restrict__ cb, float* __restrict__ nrm, int dim) {
    __shared__ float sm[256];
    const int row = blockIdx.x;
    const float* r = cb + (size_t)row * dim;
    float s = 0.f;
    for (int c = threadIdx.x; c < dim; c += 256) { float v = r[c]; s += v * v; }
    float tot = block_reduce256(s, sm);
    if (threadIdx.x == 0) nrm[row] = tot;
}

__global__ __launch_bounds__(256) void cvt_kernel(
    const float* __restrict__ s, _Float16* __restrict__ d, size_t n) {
    size_t stride = (size_t)gridDim.x * blockDim.x;
    for (size_t i = (size_t)blockIdx.x * blockDim.x + threadIdx.x; i < n; i += stride)
        d[i] = (_Float16)s[i];
}

__global__ __launch_bounds__(256) void cvt_rowscale_kernel(
    const float* __restrict__ s, _Float16* __restrict__ d,
    const float* __restrict__ rs, int cols, size_t n) {
    size_t stride = (size_t)gridDim.x * blockDim.x;
    for (size_t i = (size_t)blockIdx.x * blockDim.x + threadIdx.x; i < n; i += stride)
        d[i] = (_Float16)(s[i] * rs[i / (size_t)cols]);
}

// ---------------- WMMA GEMM: out = act(A[M,K] @ W[N,K]^T + bias) ----------------
// 256 threads = 8 waves; block tile = 128 rows x 32 cols per iteration; each
// wave owns 16 rows, computes two 16x16 N-tiles per pass so each A fragment
// feeds two WMMAs. The 32xK weight tile is staged in LDS via async loads and
// shared by all 8 waves.
template <int ACT, bool WF32, bool WF16>
__global__ __launch_bounds__(256) void wmma_gemm_kernel(
    const _Float16* __restrict__ A, const _Float16* __restrict__ W,
    const float* __restrict__ bias, float* __restrict__ outF,
    _Float16* __restrict__ outH, int M, int N, int K) {
    __shared__ _Float16 Wl[32 * 512];  // up to K=512 (32 KB)
    const int wave = threadIdx.x >> 5;
    const int lane = threadIdx.x & 31;
    const int m0 = blockIdx.x * 128 + wave * 16;
    const int ln  = lane & 15;
    const int kb  = (lane >> 4) * 8;
    const _Float16* arow = A + (size_t)(m0 + ln) * K;

    for (int n0 = 0; n0 < N; n0 += 32) {
        // async stage of W[n0:n0+32, 0:K] -> LDS (16B per lane per iter)
        const int nchunks = 4 * K;
        for (int t = threadIdx.x; t < nchunks; t += 256) {
            int idx = t * 8;
            int r = idx / K, c = idx % K;
            async_ld_lds_b128((unsigned)(idx * 2), &W[(size_t)(n0 + r) * K + c]);
        }
        wait_async0();
        __syncthreads();

        v8f acc0, acc1;
#pragma unroll
        for (int i = 0; i < 8; ++i) { acc0[i] = 0.f; acc1[i] = 0.f; }

        const _Float16* b0 = &Wl[(size_t)ln * K];
        const _Float16* b1 = &Wl[(size_t)(16 + ln) * K];
        for (int k = 0; k < K; k += 32) {
            v16h a  = frag_ld(arow + k, kb);
            v16h f0 = frag_ld(b0 + k, kb);
            v16h f1 = frag_ld(b1 + k, kb);
            acc0 = wmma_f16(a, f0, acc0);
            acc1 = wmma_f16(a, f1, acc1);
        }

        // epilogue: lane holds N = n0+ln (acc0) and n0+16+ln (acc1),
        // rows m0+(lane>>4)*8+v
        const int mb = (lane >> 4) * 8;
#pragma unroll
        for (int half = 0; half < 2; ++half) {
            const int n = n0 + half * 16 + ln;
            const float bv = bias[n];
#pragma unroll
            for (int v = 0; v < 8; ++v) {
                int m = m0 + mb + v;
                float val = (half ? acc1[v] : acc0[v]) + bv;
                if (ACT == 1) val = gelu_erf(val);
                if (ACT == 2) val = sigm(val);
                if (WF32) outF[(size_t)m * N + n] = val;
                if (WF16) outH[(size_t)m * N + n] = (_Float16)val;
            }
        }
        __syncthreads();
    }
}

// ---------------- WMMA quantize: argmin_c ( ||cb_c||^2 - 2 z.cb_c ) ----------------
// KT is the (compile-time) latent dim; each lane keeps its whole A slice in
// registers (KT/32 v16h fragments), so the code loop is pure LDS + WMMA.
template <int KT>
__global__ __launch_bounds__(256) void wmma_quant_kernel(
    const _Float16* __restrict__ Z, const _Float16* __restrict__ CB,
    const float* __restrict__ cbn, int* __restrict__ idxOut, int NC) {
    __shared__ _Float16 Wl[32 * KT];  // <= 16 KB
    const int wave = threadIdx.x >> 5;
    const int lane = threadIdx.x & 31;
    const int m0 = blockIdx.x * 128 + wave * 16;
    const int ln  = lane & 15;
    const int kb  = (lane >> 4) * 8;
    const _Float16* arow = Z + (size_t)(m0 + ln) * KT;

    v16h afr[KT / 32];
#pragma unroll
    for (int k = 0; k < KT / 32; ++k) afr[k] = frag_ld(arow + k * 32, kb);

    float bmin[8];
    int   bidx[8];
#pragma unroll
    for (int v = 0; v < 8; ++v) { bmin[v] = 3.4e38f; bidx[v] = 0; }

    for (int c0 = 0; c0 < NC; c0 += 32) {
        const int nchunks = 4 * KT;
        for (int t = threadIdx.x; t < nchunks; t += 256) {
            int idx = t * 8;
            int r = idx / KT, c = idx % KT;
            async_ld_lds_b128((unsigned)(idx * 2), &CB[(size_t)(c0 + r) * KT + c]);
        }
        wait_async0();
        __syncthreads();

        v8f acc0, acc1;
#pragma unroll
        for (int i = 0; i < 8; ++i) { acc0[i] = 0.f; acc1[i] = 0.f; }

        const _Float16* b0 = &Wl[(size_t)ln * KT];
        const _Float16* b1 = &Wl[(size_t)(16 + ln) * KT];
#pragma unroll
        for (int k = 0; k < KT / 32; ++k) {
            v16h f0 = frag_ld(b0 + k * 32, kb);
            v16h f1 = frag_ld(b1 + k * 32, kb);
            acc0 = wmma_f16(afr[k], f0, acc0);
            acc1 = wmma_f16(afr[k], f1, acc1);
        }

        const int cA = c0 + ln;
        const int cBc = c0 + 16 + ln;
        const float nA = cbn[cA];
        const float nB = cbn[cBc];
#pragma unroll
        for (int v = 0; v < 8; ++v) {
            float s0 = nA - 2.0f * acc0[v];
            if (s0 < bmin[v]) { bmin[v] = s0; bidx[v] = cA; }
            float s1 = nB - 2.0f * acc1[v];
            if (s1 < bmin[v]) { bmin[v] = s1; bidx[v] = cBc; }
        }
        __syncthreads();
    }

    // min-reduce across the 16 lanes that share the same 8 M-rows
#pragma unroll
    for (int off = 1; off < 16; off <<= 1) {
#pragma unroll
        for (int v = 0; v < 8; ++v) {
            float om = __shfl_xor(bmin[v], off, 32);
            int   oi = __shfl_xor(bidx[v], off, 32);
            if (om < bmin[v] || (om == bmin[v] && oi < bidx[v])) {
                bmin[v] = om; bidx[v] = oi;
            }
        }
    }
    if ((lane & 15) == 0) {
        const int mb = (lane >> 4) * 8;
#pragma unroll
        for (int v = 0; v < 8; ++v) idxOut[m0 + mb + v] = bidx[v];
    }
}

// ---------------- gather z_q = codebook[idx]; accumulate ||z_e - z_q||^2 ----------------
__global__ __launch_bounds__(256) void gather_mse_kernel(
    const int* __restrict__ idx, const float* __restrict__ cb,
    const float* __restrict__ ze, float* __restrict__ outQ,
    _Float16* __restrict__ outQh, float* __restrict__ idxF,
    float* __restrict__ accum, int B, int D) {
    __shared__ float sm[256];
    size_t total = (size_t)B * D;
    size_t stride = (size_t)gridDim.x * blockDim.x;
    float local = 0.f;
    for (size_t i = (size_t)blockIdx.x * blockDim.x + threadIdx.x; i < total; i += stride) {
        int r = (int)(i / (size_t)D);
        int c = (int)(i % (size_t)D);
        int code = idx[r];
        float qv = cb[(size_t)code * D + c];
        outQ[i]  = qv;
        outQh[i] = (_Float16)qv;
        float d = ze[i] - qv;
        local += d * d;
        if (c == 0) idxF[r] = (float)code;
    }
    float tot = block_reduce256(local, sm);
    if (threadIdx.x == 0) atomicAdd(accum, tot);
}

__global__ __launch_bounds__(256) void mse_sum_kernel(
    const float* __restrict__ a, const float* __restrict__ b,
    float* __restrict__ accum, size_t n) {
    __shared__ float sm[256];
    size_t stride = (size_t)gridDim.x * blockDim.x;
    float local = 0.f;
    for (size_t i = (size_t)blockIdx.x * blockDim.x + threadIdx.x; i < n; i += stride) {
        float d = a[i] - b[i];
        local += d * d;
    }
    float tot = block_reduce256(local, sm);
    if (threadIdx.x == 0) atomicAdd(accum, tot);
}

__global__ void finalize_kernel(const float* __restrict__ accum, float* __restrict__ out) {
    float cz = accum[0] / (float)((size_t)B_ * ZD);
    float cq = accum[1] / (float)((size_t)B_ * QD);
    float rl = accum[2] / (float)((size_t)B_ * FEAT);
    out[0] = rl + 0.25f * (cz + cz) + 0.25f * (cq + cq);
    out[1] = rl;
    out[2] = cz;  // commit_z
    out[3] = cz;  // codebook_z (forward-identical)
    out[4] = cq;  // commit_q
    out[5] = cq;  // codebook_q
}

// ---------------- host launcher ----------------

extern "C" void kernel_launch(void* const* d_in, const int* in_sizes, int n_in,
                              void* d_out, int out_size, void* d_ws, size_t ws_size,
                              hipStream_t stream) {
    (void)in_sizes; (void)n_in; (void)out_size; (void)ws_size;

    const float* x   = (const float*)d_in[0];
    const float* eW1 = (const float*)d_in[1];
    const float* eb1 = (const float*)d_in[2];
    const float* eW2 = (const float*)d_in[3];
    const float* eb2 = (const float*)d_in[4];
    const float* zW  = (const float*)d_in[5];
    const float* zb  = (const float*)d_in[6];
    const float* zci = (const float*)d_in[7];
    const float* zcb = (const float*)d_in[8];
    const float* qW  = (const float*)d_in[9];
    const float* qb  = (const float*)d_in[10];
    const float* qci = (const float*)d_in[11];
    const float* qcb = (const float*)d_in[12];
    const float* dW1 = (const float*)d_in[13];
    const float* db1 = (const float*)d_in[14];
    const float* dW2 = (const float*)d_in[15];
    const float* db2 = (const float*)d_in[16];
    const float* oW  = (const float*)d_in[17];
    const float* ob  = (const float*)d_in[18];

    float* out    = (float*)d_out;
    float* o_xrec = out + 6;
    float* o_zq   = o_xrec + (size_t)B_ * FEAT;
    float* o_qq   = o_zq + (size_t)B_ * ZD;
    float* o_zidx = o_qq + (size_t)B_ * QD;
    float* o_qidx = o_zidx + B_;

    char* ws = (char*)d_ws;
    size_t off = 0;
    auto alloc = [&](size_t bytes) -> char* {
        char* p = ws + off;
        off = (off + bytes + 255) & ~(size_t)255;
        return p;
    };

    float*    accum = (float*)alloc(4 * sizeof(float));
    float*    sclz  = (float*)alloc(ZD * sizeof(float));
    float*    sclq  = (float*)alloc(QD * sizeof(float));
    float*    zcbn  = (float*)alloc(NZC * sizeof(float));
    float*    qcbn  = (float*)alloc(NQC * sizeof(float));
    _Float16* xh    = (_Float16*)alloc((size_t)B_ * FEAT * 2);
    _Float16* eW1h  = (_Float16*)alloc((size_t)BOT * FEAT * 2);
    _Float16* eW2h  = (_Float16*)alloc((size_t)HID * BOT * 2);
    _Float16* zWh   = (_Float16*)alloc((size_t)ZD * HID * 2);
    _Float16* qWh   = (_Float16*)alloc((size_t)QD * ZD * 2);
    _Float16* dW1h  = (_Float16*)alloc((size_t)BOT * QD * 2);
    _Float16* dW2h  = (_Float16*)alloc((size_t)HID * BOT * 2);
    _Float16* oWh   = (_Float16*)alloc((size_t)FEAT * HID * 2);
    _Float16* zcbh  = (_Float16*)alloc((size_t)NZC * ZD * 2);
    _Float16* qcbh  = (_Float16*)alloc((size_t)NQC * QD * 2);
    _Float16* h1h   = (_Float16*)alloc((size_t)B_ * BOT * 2);
    _Float16* hh    = (_Float16*)alloc((size_t)B_ * HID * 2);
    float*    zeF   = (float*)alloc((size_t)B_ * ZD * 4);
    _Float16* zeh   = (_Float16*)alloc((size_t)B_ * ZD * 2);
    int*      zidx  = (int*)alloc((size_t)B_ * 4);
    _Float16* zqh   = (_Float16*)alloc((size_t)B_ * ZD * 2);
    float*    qeF   = (float*)alloc((size_t)B_ * QD * 4);
    _Float16* qeh   = (_Float16*)alloc((size_t)B_ * QD * 2);
    int*      qidx  = (int*)alloc((size_t)B_ * 4);
    _Float16* qqh   = (_Float16*)alloc((size_t)B_ * QD * 2);
    _Float16* d1h   = (_Float16*)alloc((size_t)B_ * BOT * 2);
    _Float16* dhh   = (_Float16*)alloc((size_t)B_ * HID * 2);

    // prep
    zero4_kernel<<<1, 32, 0, stream>>>(accum);
    lip_scale_kernel<<<ZD, 256, 0, stream>>>(zW, zci, sclz, HID);
    lip_scale_kernel<<<QD, 256, 0, stream>>>(qW, qci, sclq, ZD);
    cvt_kernel<<<2048, 256, 0, stream>>>(x, xh, (size_t)B_ * FEAT);
    cvt_kernel<<<64, 256, 0, stream>>>(eW1, eW1h, (size_t)BOT * FEAT);
    cvt_kernel<<<64, 256, 0, stream>>>(eW2, eW2h, (size_t)HID * BOT);
    cvt_rowscale_kernel<<<128, 256, 0, stream>>>(zW, zWh, sclz, HID, (size_t)ZD * HID);
    cvt_rowscale_kernel<<<64, 256, 0, stream>>>(qW, qWh, sclq, ZD, (size_t)QD * ZD);
    cvt_kernel<<<32, 256, 0, stream>>>(dW1, dW1h, (size_t)BOT * QD);
    cvt_kernel<<<64, 256, 0, stream>>>(dW2, dW2h, (size_t)HID * BOT);
    cvt_kernel<<<256, 256, 0, stream>>>(oW, oWh, (size_t)FEAT * HID);
    cvt_kernel<<<1024, 256, 0, stream>>>(zcb, zcbh, (size_t)NZC * ZD);
    cvt_kernel<<<256, 256, 0, stream>>>(qcb, qcbh, (size_t)NQC * QD);
    cb_norm_kernel<<<NZC, 256, 0, stream>>>(zcb, zcbn, ZD);
    cb_norm_kernel<<<NQC, 256, 0, stream>>>(qcb, qcbn, QD);

    dim3 gm(B_ / 128);
    // encoder
    wmma_gemm_kernel<1, false, true><<<gm, 256, 0, stream>>>(xh, eW1h, eb1, (float*)nullptr, h1h, B_, BOT, FEAT);
    wmma_gemm_kernel<1, false, true><<<gm, 256, 0, stream>>>(h1h, eW2h, eb2, (float*)nullptr, hh, B_, HID, BOT);
    // z latent + quantize
    wmma_gemm_kernel<2, true, true><<<gm, 256, 0, stream>>>(hh, zWh, zb, zeF, zeh, B_, ZD, HID);
    wmma_quant_kernel<ZD><<<gm, 256, 0, stream>>>(zeh, zcbh, zcbn, zidx, NZC);
    gather_mse_kernel<<<1024, 256, 0, stream>>>(zidx, zcb, zeF, o_zq, zqh, o_zidx, accum + 0, B_, ZD);
    // q latent + quantize
    wmma_gemm_kernel<2, true, true><<<gm, 256, 0, stream>>>(zqh, qWh, qb, qeF, qeh, B_, QD, ZD);
    wmma_quant_kernel<QD><<<gm, 256, 0, stream>>>(qeh, qcbh, qcbn, qidx, NQC);
    gather_mse_kernel<<<1024, 256, 0, stream>>>(qidx, qcb, qeF, o_qq, qqh, o_qidx, accum + 1, B_, QD);
    // decoder
    wmma_gemm_kernel<1, false, true><<<gm, 256, 0, stream>>>(qqh, dW1h, db1, (float*)nullptr, d1h, B_, BOT, QD);
    wmma_gemm_kernel<1, false, true><<<gm, 256, 0, stream>>>(d1h, dW2h, db2, (float*)nullptr, dhh, B_, HID, BOT);
    wmma_gemm_kernel<0, true, false><<<gm, 256, 0, stream>>>(dhh, oWh, ob, o_xrec, (_Float16*)nullptr, B_, FEAT, HID);
    // losses
    mse_sum_kernel<<<1024, 256, 0, stream>>>(o_xrec, x, accum + 2, (size_t)B_ * FEAT);
    finalize_kernel<<<1, 1, 0, stream>>>(accum, out);
}
